// ResidualSelfAttention_58634893525279
// MI455X (gfx1250) — matrix operand order repaired and checked by reading, not compiled, not run on hardware
//
#include <hip/hip_runtime.h>
#include <hip/hip_bf16.h>
#include <stdint.h>

typedef __bf16 bf16_t;
typedef __attribute__((ext_vector_type(16))) __bf16 v16bf;
typedef __attribute__((ext_vector_type(8)))  __bf16 v8bf;
typedef __attribute__((ext_vector_type(8)))  float  v8f;

#define HID   128
#define HEADS 8
#define DH    16
#define EDIM  16
#define TILE_E 16

// ---------- WMMA fragment helpers (gfx1250, wave32) ----------
// A fragment (16x32 bf16, MxK): lane L holds row M=L&15; elements 0..7 are
// K = kb..kb+7 and elements 8..15 are K = kb+16..kb+23, kb = (L>>4)*8.
__device__ inline v16bf load_frag_a(const bf16_t* rowptr, int kb) {
  v8bf lo = *(const v8bf*)(rowptr + kb);
  v8bf hi = *(const v8bf*)(rowptr + kb + 16);
  v16bf r;
#pragma unroll
  for (int i = 0; i < 8; ++i) { r[i] = lo[i]; r[i + 8] = hi[i]; }
  return r;
}
// B fragment (32x16 bf16, KxN): lane L holds col N=L&15; elements 0..7 are
// K = kb..kb+7 and elements 8..15 are K = kb+16..kb+23, kb = (L>>4)*8.
// Since out = in @ W^T, B[k][n] = W[n][k] -> two 16B loads from weight row n.
__device__ inline v16bf load_frag_b(const bf16_t* wrow, int kb) {
  v8bf lo = *(const v8bf*)(wrow + kb);
  v8bf hi = *(const v8bf*)(wrow + kb + 16);
  v16bf r;
#pragma unroll
  for (int i = 0; i < 8; ++i) { r[i] = lo[i]; r[i + 8] = hi[i]; }
  return r;
}
__device__ inline v8f wmma_bf16(v16bf a, v16bf b, v8f c) {
  return __builtin_amdgcn_wmma_f32_16x16x32_bf16(false, a, false, b,
                                                 (short)0, c, false, false);
}

__device__ inline void atomic_max_f32(float* addr, float val) {
  int* ia = (int*)addr;
  int old = __float_as_int(*addr);
  while (__int_as_float(old) < val) {
    int assumed = old;
    old = atomicCAS(ia, assumed, __float_as_int(val));
    if (old == assumed) break;
  }
}

// ---------- 1. LayerNorm(x) -> xn (f32 + bf16) ----------
__global__ __launch_bounds__(HID) void ln_x_kernel(
    const float* __restrict__ x, const float* __restrict__ w,
    const float* __restrict__ b, float* __restrict__ xn,
    bf16_t* __restrict__ xn_bf) {
  __shared__ float red[4];
  int n = blockIdx.x, t = threadIdx.x;
  float v = x[(size_t)n * HID + t];
  float s = v;
#pragma unroll
  for (int o = 16; o > 0; o >>= 1) s += __shfl_xor(s, o, 32);
  if ((t & 31) == 0) red[t >> 5] = s;
  __syncthreads();
  float mu = (red[0] + red[1] + red[2] + red[3]) * (1.0f / HID);
  __syncthreads();
  float d = v - mu;
  s = d * d;
#pragma unroll
  for (int o = 16; o > 0; o >>= 1) s += __shfl_xor(s, o, 32);
  if ((t & 31) == 0) red[t >> 5] = s;
  __syncthreads();
  float var = (red[0] + red[1] + red[2] + red[3]) * (1.0f / HID);
  float out = d * rsqrtf(var + 1e-5f) * w[t] + b[t];
  xn[(size_t)n * HID + t] = out;
  xn_bf[(size_t)n * HID + t] = (bf16_t)out;
}

// ---------- 2. f32 -> bf16 weight conversion ----------
__global__ void cvt_kernel(const float* __restrict__ src,
                           bf16_t* __restrict__ dst, int n) {
  int i = blockIdx.x * blockDim.x + threadIdx.x;
  if (i < n) dst[i] = (bf16_t)src[i];
}

// ---------- 3. init amax/den/m ----------
__global__ void init_kernel(float* amax, float* den, float* m, int N_) {
  int i = blockIdx.x * blockDim.x + threadIdx.x;
  if (i < N_ * HEADS) { amax[i] = -3.0e38f; den[i] = 0.0f; }
  if (i < N_ * HID) m[i] = 0.0f;
}

// ---------- 4. fused edge kernel: rel MLP + q/k/v WMMA + logits ----------
__global__ __launch_bounds__(256) void edge_kernel(
    const float* __restrict__ xn, const bf16_t* __restrict__ xn_bf,
    const float* __restrict__ p, const int* __restrict__ edge_index,
    const float* __restrict__ edge_attr,
    const float* __restrict__ ln_e_w, const float* __restrict__ ln_e_b,
    const float* __restrict__ ln_e2_w, const float* __restrict__ ln_e2_b,
    const float* __restrict__ W_edge, const float* __restrict__ We1,
    const float* __restrict__ be1, const float* __restrict__ be2,
    const bf16_t* __restrict__ We2_bf, const bf16_t* __restrict__ Wq_bf,
    const bf16_t* __restrict__ Wk_bf, const bf16_t* __restrict__ Wv_bf,
    float* __restrict__ alpha_buf, bf16_t* __restrict__ vbuf,
    float* __restrict__ amax, int E_) {
  __shared__ int   s_src[TILE_E], s_dst[TILE_E];
  __shared__ float s_rel3[TILE_E][3];
  __shared__ float s_ea2[TILE_E][EDIM];
  __shared__ float s_ebias[TILE_E][HEADS];
  __shared__ __align__(16) bf16_t s_relh[TILE_E][HID];
  __shared__ __align__(16) bf16_t s_kvin[TILE_E][HID];

  int t = threadIdx.x;
  int lane = t & 31;
  int wave = t >> 5;              // 0..7 == head index (DH==16)
  int e0 = blockIdx.x * TILE_E;

  // ---- stage A: per-edge scalar prep (16 threads) ----
  if (t < TILE_E) {
    int e = min(e0 + t, E_ - 1);
    int sI = edge_index[e];
    int dI = edge_index[E_ + e];
    s_src[t] = sI; s_dst[t] = dI;
    float pi0 = p[dI * 3], pi1 = p[dI * 3 + 1], pi2 = p[dI * 3 + 2];
    float pj0 = p[sI * 3], pj1 = p[sI * 3 + 1], pj2 = p[sI * 3 + 2];
    s_rel3[t][0] = pj0 - pi0;
    s_rel3[t][1] = pi1 * pj1 + pi2 * pj2;
    s_rel3[t][2] = -pi2 * pj1 + pi1 * pj2;
    // double layernorm of edge_attr row
    float v[EDIM]; float mu = 0.0f;
#pragma unroll
    for (int i = 0; i < EDIM; ++i) { v[i] = edge_attr[(size_t)e * EDIM + i]; mu += v[i]; }
    mu *= (1.0f / EDIM); float var = 0.0f;
#pragma unroll
    for (int i = 0; i < EDIM; ++i) { float d = v[i] - mu; var += d * d; }
    float inv = rsqrtf(var * (1.0f / EDIM) + 1e-5f);
#pragma unroll
    for (int i = 0; i < EDIM; ++i) v[i] = (v[i] - mu) * inv * ln_e_w[i] + ln_e_b[i];
    mu = 0.0f;
#pragma unroll
    for (int i = 0; i < EDIM; ++i) mu += v[i];
    mu *= (1.0f / EDIM); var = 0.0f;
#pragma unroll
    for (int i = 0; i < EDIM; ++i) { float d = v[i] - mu; var += d * d; }
    inv = rsqrtf(var * (1.0f / EDIM) + 1e-5f);
#pragma unroll
    for (int i = 0; i < EDIM; ++i)
      s_ea2[t][i] = (v[i] - mu) * inv * ln_e2_w[i] + ln_e2_b[i];
  }
  __syncthreads();

  // edge bias: 16 edges x 8 heads
  if (t < TILE_E * HEADS) {
    int r = t >> 3, h = t & 7;
    float acc = 0.0f;
#pragma unroll
    for (int i = 0; i < EDIM; ++i) acc += s_ea2[r][i] * W_edge[h * EDIM + i];
    s_ebias[r][h] = acc;
  }
  // rel_h = leaky(rel3 @ We1^T + be1), 16x128, all 256 threads (8 each)
  {
    int r = t >> 4;
    int jb = (t & 15) * 8;
    float r0 = s_rel3[r][0], r1 = s_rel3[r][1], r2 = s_rel3[r][2];
#pragma unroll
    for (int jj = 0; jj < 8; ++jj) {
      int j = jb + jj;
      float h = We1[j * 3] * r0 + We1[j * 3 + 1] * r1 + We1[j * 3 + 2] * r2 + be1[j];
      h = (h >= 0.0f) ? h : 0.01f * h;
      s_relh[r][j] = (bf16_t)h;
    }
  }
  __syncthreads();

  int n0   = wave * DH;            // this wave's 16 output columns
  int colL = n0 + (lane & 15);
  int kbA  = (lane >> 4) * 8;
  int kbB  = (lane >> 4) * 8;
  int rhal = (lane >> 4) * 8;      // D-frag row base for this lane

  // ---- stage B: rel2 = relh @ We2^T ; kv_in = x_j + rel2 + be2 ----
  {
    v8f acc = {};
    const bf16_t* arow = &s_relh[lane & 15][0];
    const bf16_t* brow = We2_bf + (size_t)colL * HID;
#pragma unroll
    for (int k0 = 0; k0 < HID; k0 += 32)
      acc = wmma_bf16(load_frag_a(arow, k0 + kbA), load_frag_b(brow, k0 + kbB), acc);
    float bias = be2[colL];
#pragma unroll
    for (int v = 0; v < 8; ++v) {
      int r = rhal + v;
      float kv = acc[v] + bias + xn[(size_t)s_src[r] * HID + colL];
      s_kvin[r][colL] = (bf16_t)kv;
    }
  }
  __syncthreads();

  // ---- stage C: q = x_i @ Wq^T ; k,v = kv_in @ {Wk,Wv}^T ----
  v8f accq = {}, acck = {}, accv = {};
  {
    const bf16_t* arow_q = xn_bf + (size_t)s_dst[lane & 15] * HID;
    const bf16_t* arow_k = &s_kvin[lane & 15][0];
    const bf16_t* bq = Wq_bf + (size_t)colL * HID;
    const bf16_t* bk = Wk_bf + (size_t)colL * HID;
    const bf16_t* bv = Wv_bf + (size_t)colL * HID;
#pragma unroll
    for (int k0 = 0; k0 < HID; k0 += 32) {
      v16bf aq = load_frag_a(arow_q, k0 + kbA);
      v16bf ak = load_frag_a(arow_k, k0 + kbA);
      accq = wmma_bf16(aq, load_frag_b(bq, k0 + kbB), accq);
      acck = wmma_bf16(ak, load_frag_b(bk, k0 + kbB), acck);
      accv = wmma_bf16(ak, load_frag_b(bv, k0 + kbB), accv);
    }
  }

  // ---- stage D: alpha = sum_cols(q*k)/4 + ebias ; store v, alpha, amax ----
  float sums[8];
#pragma unroll
  for (int v = 0; v < 8; ++v) {
    float pr = accq[v] * acck[v];
    pr += __shfl_xor(pr, 1, 32);
    pr += __shfl_xor(pr, 2, 32);
    pr += __shfl_xor(pr, 4, 32);
    pr += __shfl_xor(pr, 8, 32);
    sums[v] = pr;
  }
#pragma unroll
  for (int v = 0; v < 8; ++v) {
    int r = rhal + v;
    if (e0 + r < E_)
      vbuf[(size_t)(e0 + r) * HID + colL] = (bf16_t)accv[v];
  }
  if ((lane & 15) == 0) {
#pragma unroll
    for (int v = 0; v < 8; ++v) {
      int r = rhal + v;
      if (e0 + r < E_) {
        float alpha = sums[v] * 0.25f + s_ebias[r][wave];
        alpha_buf[(size_t)(e0 + r) * HEADS + wave] = alpha;
        atomic_max_f32(&amax[(size_t)s_dst[r] * HEADS + wave], alpha);
      }
    }
  }
}

// ---------- 5. exp + denominator ----------
__global__ void softmax_kernel(const int* __restrict__ edge_index,
                               float* __restrict__ alpha_ex,
                               const float* __restrict__ amax,
                               float* __restrict__ den, int E_) {
  int i = blockIdx.x * blockDim.x + threadIdx.x;
  if (i >= E_ * HEADS) return;
  int e = i >> 3, h = i & 7;
  int d = edge_index[E_ + e];
  float ex = expf(alpha_ex[i] - amax[(size_t)d * HEADS + h]);
  alpha_ex[i] = ex;
  atomicAdd(&den[(size_t)d * HEADS + h], ex);
}

// ---------- 6. weighted aggregation m = segsum(v * a) ----------
__global__ __launch_bounds__(HID) void aggregate_kernel(
    const int* __restrict__ edge_index, const float* __restrict__ exbuf,
    const float* __restrict__ den, const bf16_t* __restrict__ vbuf,
    float* __restrict__ m, int E_) {
  int e = blockIdx.x;
  int c = threadIdx.x;
  int h = c >> 4;
  int d = edge_index[E_ + e];
  float a = exbuf[(size_t)e * HEADS + h] /
            (den[(size_t)d * HEADS + h] + 1e-16f);
  float v = (float)vbuf[(size_t)e * HID + c];
  atomicAdd(&m[(size_t)d * HID + c], v * a);
}

// ---------- 7. node tail: Wd1/Wg/Wd2/Wgs + gating + geometry ----------
__global__ __launch_bounds__(HID) void node_final_kernel(
    const float* __restrict__ x, const float* __restrict__ p,
    const float* __restrict__ xn, const float* __restrict__ m,
    const float* __restrict__ Wd1, const float* __restrict__ bd1,
    const float* __restrict__ Wd2, const float* __restrict__ bd2,
    const float* __restrict__ Wg, const float* __restrict__ bg,
    const float* __restrict__ Wgs, const float* __restrict__ bgs,
    float* __restrict__ out_x, float* __restrict__ out_p) {
  __shared__ float s_m[HID], s_xn[HID], s_d1[HID], s_mx[HID], s_mp[3];
  int n = blockIdx.x, t = threadIdx.x;
  s_m[t]  = m[(size_t)n * HID + t];
  s_xn[t] = xn[(size_t)n * HID + t];
  __syncthreads();
  // d1 = leaky(m @ Wd1^T + bd1)
  float acc = bd1[t];
  const float* wd = Wd1 + (size_t)t * HID;
  for (int i = 0; i < HID; ++i) acc += s_m[i] * wd[i];
  s_d1[t] = (acc >= 0.0f) ? acc : 0.01f * acc;
  // gate = sigmoid([xn, m] @ Wg^T + bg)
  float g = bg[t];
  const float* wg = Wg + (size_t)t * (2 * HID);
  for (int i = 0; i < HID; ++i) g += s_xn[i] * wg[i];
  for (int i = 0; i < HID; ++i) g += s_m[i] * wg[HID + i];
  g = 1.0f / (1.0f + expf(-g));
  float mx = g * s_m[t];
  s_mx[t] = mx;
  out_x[(size_t)n * HID + t] = x[(size_t)n * HID + t] + mx;
  __syncthreads();
  if (t < 3) {
    float mp = bd2[t];
    const float* w2 = Wd2 + (size_t)t * HID;
    for (int i = 0; i < HID; ++i) mp += s_d1[i] * w2[i];
    float gs = bgs[t];
    const float* wgs = Wgs + (size_t)t * (2 * HID);
    for (int i = 0; i < HID; ++i) gs += s_xn[i] * wgs[i];
    for (int i = 0; i < HID; ++i) gs += s_mx[i] * wgs[HID + i];
    gs = 1.0f / (1.0f + expf(-gs));
    s_mp[t] = gs * mp;
  }
  __syncthreads();
  if (t == 0) {
    float mp0 = s_mp[0], mp1 = s_mp[1] + 1.0f, mp2 = s_mp[2];
    float nr = sqrtf(mp1 * mp1 + mp2 * mp2);
    float c2 = mp1 / (nr + 1e-8f), s2 = mp2 / (nr + 1e-8f);
    float eta = p[n * 3 + 0] + mp0;
    float pp1 = p[n * 3 + 1], pp2 = p[n * 3 + 2];
    float f1 = c2 * pp1 - s2 * pp2;
    float f2 = s2 * pp1 + c2 * pp2;
    float nr2 = sqrtf(f1 * f1 + f2 * f2);
    out_p[n * 3 + 0] = eta;
    out_p[n * 3 + 1] = f1 / (nr2 + 1e-8f);
    out_p[n * 3 + 2] = f2 / (nr2 + 1e-8f);
  }
}

extern "C" void kernel_launch(void* const* d_in, const int* in_sizes, int n_in,
                              void* d_out, int out_size, void* d_ws, size_t ws_size,
                              hipStream_t stream) {
  const float* x        = (const float*)d_in[0];
  const float* p        = (const float*)d_in[1];
  const int*   eidx     = (const int*)d_in[2];
  const float* eattr    = (const float*)d_in[3];
  const float* ln_x_w   = (const float*)d_in[4];
  const float* ln_x_b   = (const float*)d_in[5];
  const float* ln_e_w   = (const float*)d_in[6];
  const float* ln_e_b   = (const float*)d_in[7];
  const float* ln_e2_w  = (const float*)d_in[8];
  const float* ln_e2_b  = (const float*)d_in[9];
  const float* Wq       = (const float*)d_in[10];
  const float* Wk       = (const float*)d_in[11];
  const float* Wv       = (const float*)d_in[12];
  const float* W_edge   = (const float*)d_in[13];
  const float* We1      = (const float*)d_in[14];
  const float* be1      = (const float*)d_in[15];
  const float* We2      = (const float*)d_in[16];
  const float* be2      = (const float*)d_in[17];
  const float* Wd1      = (const float*)d_in[18];
  const float* bd1      = (const float*)d_in[19];
  const float* Wd2      = (const float*)d_in[20];
  const float* bd2      = (const float*)d_in[21];
  const float* Wg       = (const float*)d_in[22];
  const float* bg       = (const float*)d_in[23];
  const float* Wgs      = (const float*)d_in[24];
  const float* bgs      = (const float*)d_in[25];

  const int N = in_sizes[0] / HID;
  const int E = in_sizes[2] / 2;

  // workspace carve-up (256B aligned)
  char* ws = (char*)d_ws;
  size_t off = 0;
  auto alloc = [&](size_t bytes) {
    void* q = ws + off;
    off += (bytes + 255) & ~(size_t)255;
    return q;
  };
  float*  xn_f32 = (float*)alloc((size_t)N * HID * 4);
  bf16_t* xn_bf  = (bf16_t*)alloc((size_t)N * HID * 2);
  bf16_t* Wq_bf  = (bf16_t*)alloc((size_t)HID * HID * 2);
  bf16_t* Wk_bf  = (bf16_t*)alloc((size_t)HID * HID * 2);
  bf16_t* Wv_bf  = (bf16_t*)alloc((size_t)HID * HID * 2);
  bf16_t* We2_bf = (bf16_t*)alloc((size_t)HID * HID * 2);
  float*  alpha  = (float*)alloc((size_t)E * HEADS * 4);
  bf16_t* vbuf   = (bf16_t*)alloc((size_t)E * HID * 2);
  float*  amax   = (float*)alloc((size_t)N * HEADS * 4);
  float*  den    = (float*)alloc((size_t)N * HEADS * 4);
  float*  mbuf   = (float*)alloc((size_t)N * HID * 4);
  (void)ws_size; (void)n_in; (void)out_size;

  float* out_x = (float*)d_out;
  float* out_p = out_x + (size_t)N * HID;

  // 1. layernorm x
  ln_x_kernel<<<N, HID, 0, stream>>>(x, ln_x_w, ln_x_b, xn_f32, xn_bf);
  // 2. weight conversions
  {
    int n = HID * HID, blk = 256, g = (n + blk - 1) / blk;
    cvt_kernel<<<g, blk, 0, stream>>>(Wq, Wq_bf, n);
    cvt_kernel<<<g, blk, 0, stream>>>(Wk, Wk_bf, n);
    cvt_kernel<<<g, blk, 0, stream>>>(Wv, Wv_bf, n);
    cvt_kernel<<<g, blk, 0, stream>>>(We2, We2_bf, n);
  }
  // 3. init
  {
    int n = N * HID, blk = 256, g = (n + blk - 1) / blk;
    init_kernel<<<g, blk, 0, stream>>>(amax, den, mbuf, N);
  }
  // 4. fused edge WMMA kernel
  edge_kernel<<<(E + TILE_E - 1) / TILE_E, 256, 0, stream>>>(
      xn_f32, xn_bf, p, eidx, eattr, ln_e_w, ln_e_b, ln_e2_w, ln_e2_b,
      W_edge, We1, be1, be2, We2_bf, Wq_bf, Wk_bf, Wv_bf,
      alpha, vbuf, amax, E);
  // 5. softmax numerator / denominator
  {
    int n = E * HEADS, blk = 256, g = (n + blk - 1) / blk;
    softmax_kernel<<<g, blk, 0, stream>>>(eidx, alpha, amax, den, E);
  }
  // 6. aggregate
  aggregate_kernel<<<E, HID, 0, stream>>>(eidx, alpha, den, vbuf, mbuf, E);
  // 7. node tail
  node_final_kernel<<<N, HID, 0, stream>>>(x, p, xn_f32, mbuf,
                                           Wd1, bd1, Wd2, bd2,
                                           Wg, bg, Wgs, bgs,
                                           out_x, out_p);
}
